// WriteState_50457275794065
// MI455X (gfx1250) — matrix operand order repaired and checked by reading, not compiled, not run on hardware
//
#include <hip/hip_runtime.h>

// ---------------------------------------------------------------------------
// WriteState update for MI455X (gfx1250, wave32, WMMA + TDM).
//
//   1) pack_a:       x'[b, n*D+d] = probs[b,n]*tensor[b,d] -> bf16 hi/lo
//   2) decay_tables: sigmoid tables shared by all (b,h)
//   3) gemm_all:     C(64x20480) = x'(64x8192) @ W(8192x20480), fused 4-proj.
//                    A panels (64 x 128k, hi+lo = 32KB) are DMA'd to LDS by
//                    the Tensor Data Mover (double-buffered, wave0-issued,
//                    s_wait_tensorcnt + barrier), shared by all 8 waves ->
//                    8x less A traffic than per-wave L2 fetches. Weights are
//                    streamed once from HBM with NT f32 loads. Math is
//                    3-WMMA bf16 hi/lo split (~f32 accuracy, bf16 rate).
//   4) epilogue:     activations + decayed outer-product state update,
//                    streaming matrix 134MB in / 134MB out with NT hints.
//
// Roofline: ~0.94 GB @ 23.3 TB/s => ~40us floor; GEMM = 21.5 GFLOP at
// bf16-WMMA/3 effective rate stays below the floor => memory-bound design.
// ---------------------------------------------------------------------------

typedef __bf16 bf16_t;
typedef __attribute__((ext_vector_type(8)))  bf16_t v8bf;
typedef __attribute__((ext_vector_type(16))) bf16_t v16bf;
typedef __attribute__((ext_vector_type(8)))  float  v8f;
typedef __attribute__((ext_vector_type(4)))  float  vf4;
typedef __attribute__((ext_vector_type(4)))  unsigned int u32x4;
typedef __attribute__((ext_vector_type(8)))  int    i32x8;
typedef __attribute__((ext_vector_type(4)))  int    i32x4;

#define B_    64
#define D_    1024
#define H_    64
#define DK_   64
#define DV_   128
#define BANK_ 8
#define KD    (BANK_ * D_)           // 8192 contraction length
#define O_K   (H_ * DK_)             // 4096
#define O_V   (H_ * DV_)             // 8192
#define KP    128                    // K panel staged in LDS per block
#define NPAN  (KD / KP)              // 64 panels

#if defined(__has_builtin)
#if __has_builtin(__builtin_amdgcn_tensor_load_to_lds) && \
    __has_builtin(__builtin_amdgcn_s_wait_tensorcnt)
#define HAVE_TDM 1
#endif
#endif
#ifndef HAVE_TDM
#define HAVE_TDM 0
#endif

__device__ __forceinline__ float fast_sigmoid(float x) {
  return 1.0f / (1.0f + __expf(-x));
}

// --------------------------------------------------------------------------
// 1) Build prob-scaled activation matrix x' (64 x 8192) as bf16 hi/lo pair.
// --------------------------------------------------------------------------
__global__ __launch_bounds__(256)
void pack_a(const float* __restrict__ tensor, const float* __restrict__ probs,
            bf16_t* __restrict__ Ahi, bf16_t* __restrict__ Alo) {
  int g = blockIdx.x * 256 + threadIdx.x;
  int b = g >> 13;
  int r = g & (KD - 1);
  int n = r >> 10;
  int d = r & (D_ - 1);
  float v  = probs[b * BANK_ + n] * tensor[b * D_ + d];
  bf16_t h = (bf16_t)v;
  Ahi[g] = h;
  Alo[g] = (bf16_t)(v - (float)h);
}

// --------------------------------------------------------------------------
// 2) Decay tables: ndec[i] = sigmoid(kdl[i]); mdec[i,j] = sigmoid(vdl)*ndec[i]
// --------------------------------------------------------------------------
__global__ __launch_bounds__(256)
void decay_tables(const float* __restrict__ kdl, const float* __restrict__ vdl,
                  float* __restrict__ mdec, float* __restrict__ ndec) {
  int g = blockIdx.x * 256 + threadIdx.x;
  int i = g >> 7;
  float nd = fast_sigmoid(kdl[i]);
  mdec[g] = nd * fast_sigmoid(vdl[g]);
  if ((g & (DV_ - 1)) == 0) ndec[i] = nd;
}

#if HAVE_TDM
// --------------------------------------------------------------------------
// TDM: DMA one A panel (64 rows x KP k-cols of bf16 = 64 x 256B) into LDS.
// D# per CDNA5 ISA (2-D, 8-byte elements):
//   tile_dim0 = 32 units (256B row slice), tile_dim1 = 64 rows,
//   tensor_dim0_stride = 2048 units (16KB row pitch of the 64x8192 matrix).
// 6-arg builtin form (clang-23 / therock headers): groups 2/3 + extra group
// zero-filled (2-D tile), cpol = 0.
// --------------------------------------------------------------------------
__device__ __forceinline__ void tdm_load_panel(const bf16_t* gbase, int kbase,
                                               unsigned int lds_off) {
  unsigned long long ga =
      (unsigned long long)(uintptr_t)gbase + (unsigned long long)kbase * 2ull;
  u32x4 g0;
  g0[0] = 1u;                                   // count=1, user descriptor
  g0[1] = lds_off;                              // lds_addr (bytes)
  g0[2] = (unsigned int)ga;                     // global_addr[31:0]
  g0[3] = (unsigned int)(ga >> 32) | (2u << 30);// global_addr[56:32] | type=2
  i32x8 g1;
  g1[0] = 0x00030000;                           // data_size=3 (8B units)
  g1[1] = (int)(2048u << 16);                   // tensor_dim0 = 2048 units
  g1[2] = (int)(64u << 16);                     // tensor_dim1 = 64 rows
  g1[3] = (int)(32u << 16);                     // tile_dim0 = 32 units
  g1[4] = 64;                                   // tile_dim1 = 64, tile_dim2=0
  g1[5] = 2048;                                 // tensor_dim0_stride (units)
  g1[6] = 0;
  g1[7] = 0;
  i32x4 z4 = {0, 0, 0, 0};
  i32x8 z8 = {0, 0, 0, 0, 0, 0, 0, 0};
  __builtin_amdgcn_tensor_load_to_lds(g0, g1, z4, z4, z8, 0);
}
#endif

// --------------------------------------------------------------------------
// 3) Fused GEMM. Block -> (projection, 128-wide N tile); 8 waves/block, each
//    wave owns a 64x16 strip (4 m-tiles). A panels live in LDS (TDM-fed).
// --------------------------------------------------------------------------
__global__ __launch_bounds__(256)
void gemm_all(const bf16_t* __restrict__ Ahi, const bf16_t* __restrict__ Alo,
              const float* __restrict__ kW, const float* __restrict__ vW,
              const float* __restrict__ wW, const float* __restrict__ eW,
              float* __restrict__ Kp, float* __restrict__ Vp,
              float* __restrict__ Wp, float* __restrict__ Ep) {
  const int bid = blockIdx.x;                  // 0..159
  const float* Wt; float* C; int O; int nt;
  if (bid < 32)       { Wt = kW; C = Kp; O = O_K; nt = bid; }
  else if (bid < 96)  { Wt = vW; C = Vp; O = O_V; nt = bid - 32; }
  else if (bid < 128) { Wt = wW; C = Wp; O = O_K; nt = bid - 96; }
  else                { Wt = eW; C = Ep; O = O_K; nt = bid - 128; }

  const int lane = threadIdx.x & 31;
  const int wave = threadIdx.x >> 5;
  const int lr   = lane & 15;
  const int lh   = lane >> 4;
  const int n0   = nt * 128 + wave * 16;

  v8f acc[4] = {v8f{}, v8f{}, v8f{}, v8f{}};

  // B: this lane supplies column n0+lr, k = 8*lh + {0..7, 16..23} of each 32-k tile.
  const float* wrow = Wt + (size_t)(8 * lh) * O + n0 + lr;

#if HAVE_TDM
  __shared__ bf16_t sA[2][2][64 * KP];         // [buf][hi/lo][row*KP+k] 64KB
  const bool w0 = (threadIdx.x < 32);          // wave-uniform; TDM ignores EXEC
  if (w0) {
    tdm_load_panel(Ahi, 0, (unsigned int)(uintptr_t)&sA[0][0][0]);
    tdm_load_panel(Alo, 0, (unsigned int)(uintptr_t)&sA[0][1][0]);
  }

  for (int p = 0; p < NPAN; ++p) {
    const int buf = p & 1;

    // ---- prefetch this panel's B slices (64 x NT f32 loads, in flight across
    //      the barrier / TDM wait)
    vf4 bq[4][4];
#pragma unroll
    for (int ks = 0; ks < 4; ++ks)
#pragma unroll
      for (int q = 0; q < 4; ++q)
#pragma unroll
        for (int e = 0; e < 4; ++e)
          bq[ks][q][e] = __builtin_nontemporal_load(
              wrow + (size_t)(ks * 32 + 4 * q + 8 * (q >> 1) + e) * O);
    wrow += (size_t)KP * O;

    // ---- wave0: kick TDM for panel p+1, then wait for panel p's 2 ops
    if (w0) {
      if (p + 1 < NPAN) {
        const int nb = (p + 1) & 1;
        tdm_load_panel(Ahi, (p + 1) * KP, (unsigned int)(uintptr_t)&sA[nb][0][0]);
        tdm_load_panel(Alo, (p + 1) * KP, (unsigned int)(uintptr_t)&sA[nb][1][0]);
        __builtin_amdgcn_s_wait_tensorcnt(2);
      } else {
        __builtin_amdgcn_s_wait_tensorcnt(0);
      }
    }
    __syncthreads();                           // panel p visible to all waves

    // ---- compute 4 k-steps from LDS panel
#pragma unroll
    for (int ks = 0; ks < 4; ++ks) {
      v16bf Bhi, Blo;
#pragma unroll
      for (int q = 0; q < 4; ++q)
#pragma unroll
        for (int e = 0; e < 4; ++e) {
          float  x = bq[ks][q][e];
          bf16_t h = (bf16_t)x;
          Bhi[q * 4 + e] = h;
          Blo[q * 4 + e] = (bf16_t)(x - (float)h);
        }

      const int col = ks * 32 + lh * 8;
#pragma unroll
      for (int t = 0; t < 4; ++t) {
        const bf16_t* ph = &sA[buf][0][(t * 16 + lr) * KP + col];
        const bf16_t* pl = &sA[buf][1][(t * 16 + lr) * KP + col];
        v8bf h0 = *(const v8bf*)(ph);
        v8bf h1 = *(const v8bf*)(ph + 16);
        v8bf l0 = *(const v8bf*)(pl);
        v8bf l1 = *(const v8bf*)(pl + 16);
        v16bf Af = __builtin_shufflevector(h0, h1, 0,1,2,3,4,5,6,7,8,9,10,11,12,13,14,15);
        v16bf Al = __builtin_shufflevector(l0, l1, 0,1,2,3,4,5,6,7,8,9,10,11,12,13,14,15);

        acc[t] = __builtin_amdgcn_wmma_f32_16x16x32_bf16(
            false, Af, false, Bhi, (short)0, acc[t], false, false);
        acc[t] = __builtin_amdgcn_wmma_f32_16x16x32_bf16(
            false, Af, false, Blo, (short)0, acc[t], false, false);
        acc[t] = __builtin_amdgcn_wmma_f32_16x16x32_bf16(
            false, Al, false, Bhi, (short)0, acc[t], false, false);
      }
    }
    __syncthreads();                           // done reading buf -> reusable
  }
#else
  // Fallback (no TDM builtin): per-wave A fetches straight from L2.
  const bf16_t* ah = Ahi + (size_t)lr * KD + lh * 8;
  const bf16_t* al = Alo + (size_t)lr * KD + lh * 8;
  for (int kb = 0; kb < KD; kb += 32) {
    vf4 fq[4];
#pragma unroll
    for (int q = 0; q < 4; ++q)
#pragma unroll
      for (int e = 0; e < 4; ++e)
        fq[q][e] = __builtin_nontemporal_load(
            wrow + (size_t)(4 * q + 8 * (q >> 1) + e) * O);
    wrow += (size_t)32 * O;
    v16bf Bhi, Blo;
#pragma unroll
    for (int q = 0; q < 4; ++q)
#pragma unroll
      for (int e = 0; e < 4; ++e) {
        float  x = fq[q][e];
        bf16_t h = (bf16_t)x;
        Bhi[q * 4 + e] = h;
        Blo[q * 4 + e] = (bf16_t)(x - (float)h);
      }
#pragma unroll
    for (int t = 0; t < 4; ++t) {
      const bf16_t* pa = ah + (size_t)t * 16 * KD + kb;
      const bf16_t* pl = al + (size_t)t * 16 * KD + kb;
      v8bf h0 = *(const v8bf*)(pa);
      v8bf h1 = *(const v8bf*)(pa + 16);
      v8bf l0 = *(const v8bf*)(pl);
      v8bf l1 = *(const v8bf*)(pl + 16);
      v16bf Af = __builtin_shufflevector(h0, h1, 0,1,2,3,4,5,6,7,8,9,10,11,12,13,14,15);
      v16bf Al = __builtin_shufflevector(l0, l1, 0,1,2,3,4,5,6,7,8,9,10,11,12,13,14,15);
      acc[t] = __builtin_amdgcn_wmma_f32_16x16x32_bf16(
          false, Af, false, Bhi, (short)0, acc[t], false, false);
      acc[t] = __builtin_amdgcn_wmma_f32_16x16x32_bf16(
          false, Af, false, Blo, (short)0, acc[t], false, false);
      acc[t] = __builtin_amdgcn_wmma_f32_16x16x32_bf16(
          false, Al, false, Bhi, (short)0, acc[t], false, false);
    }
  }
#endif

  // ---- store: D layout m = t*16 + 8*lh + r, n = n0 + lr
#pragma unroll
  for (int t = 0; t < 4; ++t)
#pragma unroll
    for (int r = 0; r < 8; ++r)
      C[(size_t)(t * 16 + lh * 8 + r) * O + n0 + lr] = acc[t][r];
}

// --------------------------------------------------------------------------
// 4) Epilogue: one block per (b,h). k/w/e/v scalars (+prob-weighted biases)
//    into LDS, write new_normalizer, then NT-stream the 64x128 state tile.
// --------------------------------------------------------------------------
__global__ __launch_bounds__(256)
void epilogue(const float* __restrict__ matrix, const float* __restrict__ normalizer,
              const float* __restrict__ probs,
              const float* __restrict__ key_b, const float* __restrict__ val_b,
              const float* __restrict__ write_b, const float* __restrict__ erase_b,
              const float* __restrict__ Kp, const float* __restrict__ Vp,
              const float* __restrict__ Wp, const float* __restrict__ Ep,
              const float* __restrict__ mdec, const float* __restrict__ ndec,
              float* __restrict__ out_matrix, float* __restrict__ out_norm) {
  const int bh = blockIdx.x;
  const int b  = bh >> 6;
  const int h  = bh & (H_ - 1);
  const int t  = threadIdx.x;

  __shared__ float s_k[DK_], s_w[DK_], s_e[DK_], s_v[DV_], s_p[BANK_];

  if (t < BANK_) s_p[t] = probs[b * BANK_ + t];
  __syncthreads();

  if (t < DK_) {
    const int col = h * DK_ + t;
    float bk = 0.f, bw = 0.f, be = 0.f;
#pragma unroll
    for (int n = 0; n < BANK_; ++n) {
      float p = s_p[n];
      bk = fmaf(p, key_b[n * O_K + col], bk);
      bw = fmaf(p, write_b[n * O_K + col], bw);
      be = fmaf(p, erase_b[n * O_K + col], be);
    }
    float kr = Kp[b * O_K + col] + bk;
    float kv = kr > 0.f ? kr + 1.f : __expf(kr);       // elu(x)+1
    float wv = fast_sigmoid(Wp[b * O_K + col] + bw);
    float ev = fast_sigmoid(Ep[b * O_K + col] + be);
    s_k[t] = kv; s_w[t] = wv; s_e[t] = ev;

    float nd  = ndec[t];
    float ne  = fmaxf(ev, nd);
    int   idx = b * (H_ * DK_) + h * DK_ + t;
    out_norm[idx] = normalizer[idx] * (1.f - wv * ne) + kv * nd * wv;
  }
  if (t < DV_) {
    const int col = h * DV_ + t;
    float bv = 0.f;
#pragma unroll
    for (int n = 0; n < BANK_; ++n) bv = fmaf(s_p[n], val_b[n * O_V + col], bv);
    s_v[t] = Vp[b * O_V + col] + bv;
  }
  __syncthreads();

  const size_t base = (size_t)bh * (DK_ * DV_);
#pragma unroll
  for (int rr = 0; rr < 8; ++rr) {
    int flat = t + rr * 256;
    int i    = flat >> 5;
    int j    = (flat & 31) * 4;
    size_t off = base + (size_t)i * DV_ + j;

    vf4 m  = __builtin_nontemporal_load((const vf4*)(matrix + off));
    vf4 md = *(const vf4*)(mdec + i * DV_ + j);
    float kv = s_k[i], wv = s_w[i], ev = s_e[i];
    vf4 o;
#pragma unroll
    for (int c = 0; c < 4; ++c) {
      float d = md[c];
      o[c] = m[c] * (1.f - wv * fmaxf(ev, d)) + kv * s_v[j + c] * d * wv;
    }
    __builtin_nontemporal_store(o, (vf4*)(out_matrix + off));
  }
}

// --------------------------------------------------------------------------
extern "C" void kernel_launch(void* const* d_in, const int* in_sizes, int n_in,
                              void* d_out, int out_size, void* d_ws, size_t ws_size,
                              hipStream_t stream) {
  (void)in_sizes; (void)n_in; (void)out_size; (void)ws_size;

  const float* tensor  = (const float*)d_in[0];
  const float* matrix  = (const float*)d_in[1];
  const float* normal  = (const float*)d_in[2];
  const float* probs   = (const float*)d_in[3];
  const float* key_W   = (const float*)d_in[4];
  const float* key_b   = (const float*)d_in[5];
  const float* val_W   = (const float*)d_in[6];
  const float* val_b   = (const float*)d_in[7];
  const float* write_W = (const float*)d_in[8];
  const float* write_b = (const float*)d_in[9];
  const float* erase_W = (const float*)d_in[10];
  const float* erase_b = (const float*)d_in[11];
  const float* kdl     = (const float*)d_in[12];
  const float* vdl     = (const float*)d_in[13];

  char* ws = (char*)d_ws;
  bf16_t* Ahi  = (bf16_t*)(ws);                          // 1 MB
  bf16_t* Alo  = (bf16_t*)(ws + (1u << 20));             // 1 MB
  float*  mdec = (float*)(ws + (2u << 20));              // 32 KB
  float*  ndec = (float*)(ws + (2u << 20) + 32768);      // 256 B
  float*  Kp   = (float*)(ws + (2u << 20) + 65536);      // 1 MB
  float*  Vp   = Kp + (size_t)B_ * O_K;                  // 2 MB
  float*  Wp   = Vp + (size_t)B_ * O_V;                  // 1 MB
  float*  Ep   = Wp + (size_t)B_ * O_K;                  // 1 MB

  float* out_matrix = (float*)d_out;
  float* out_norm   = out_matrix + (size_t)B_ * H_ * DK_ * DV_;

  pack_a<<<(B_ * KD) / 256, 256, 0, stream>>>(tensor, probs, Ahi, Alo);
  decay_tables<<<(DK_ * DV_) / 256, 256, 0, stream>>>(kdl, vdl, mdec, ndec);
  gemm_all<<<160, 256, 0, stream>>>(Ahi, Alo, key_W, val_W, write_W, erase_W,
                                    Kp, Vp, Wp, Ep);
  epilogue<<<B_ * H_, 256, 0, stream>>>(matrix, normal, probs,
                                        key_b, val_b, write_b, erase_b,
                                        Kp, Vp, Wp, Ep, mdec, ndec,
                                        out_matrix, out_norm);
}
